// QLoRALinear_9423158247447
// MI455X (gfx1250) — matrix operand order repaired and checked by reading, not compiled, not run on hardware
//
#include <hip/hip_runtime.h>

// ---- problem constants (QLoRA linear: [4,2048,4096] x [4096,4096], rank 16)
constexpr int Mdim = 8192;   // B*S
constexpr int Kdim = 4096;   // D_IN
constexpr int Ndim = 4096;   // D_OUT
constexpr int RANK = 16;

typedef __attribute__((ext_vector_type(2)))  float    v2f;
typedef __attribute__((ext_vector_type(4)))  float    v4f;
typedef __attribute__((ext_vector_type(4)))  int      v4i;
typedef __attribute__((ext_vector_type(8)))  _Float16 v8h;
typedef __attribute__((ext_vector_type(16))) _Float16 v16h;
typedef __attribute__((ext_vector_type(8)))  float    v8f;

union V16H { v16h v; v8h h[2]; };

// ---- CDNA5 async global->LDS path (guarded; fallback = synchronous staging)
#if defined(__has_builtin)
#  if __has_builtin(__builtin_amdgcn_global_load_async_to_lds_b128)
#    define HAVE_ASYNC_LDS 1
#  endif
#endif

#ifdef HAVE_ASYNC_LDS
   // probe-revealed signature: (int4* src, int4* dst, imm offset, imm cpol)
#  define ASYNC_CP_B128(g, l) \
     __builtin_amdgcn_global_load_async_to_lds_b128((v4i*)(g), (v4i*)(l), 0, 0)
#  if __has_builtin(__builtin_amdgcn_s_wait_asynccnt)
#    define WAIT_ASYNC(n) __builtin_amdgcn_s_wait_asynccnt(n)
#  else
#    define WAIT_ASYNC(n) asm volatile("s_wait_asynccnt %0" :: "i"(n) : "memory")
#  endif
#endif

__device__ inline v8h cvt8(v4f a, v4f b) {
  v8h r;
  r[0]=(_Float16)a[0]; r[1]=(_Float16)a[1]; r[2]=(_Float16)a[2]; r[3]=(_Float16)a[3];
  r[4]=(_Float16)b[0]; r[5]=(_Float16)b[1]; r[6]=(_Float16)b[2]; r[7]=(_Float16)b[3];
  return r;
}
__device__ inline v4f cvt4i(v4i a) {
  v4f r;
  r[0]=(float)a[0]; r[1]=(float)a[1]; r[2]=(float)a[2]; r[3]=(float)a[3];
  return r;
}

// -------------------------------------------------------------------------
// Stage 1: per 16-row tile (one wave), compute
//   R[m]   = sum_i x[m,i]                      (fp32, shfl reduction)
//   T[m,r] = sum_i x[m,i] * A[r,i]             (WMMA f32_16x16x32_f16, N=16)
// -------------------------------------------------------------------------
__global__ __launch_bounds__(256) void qlora_stage1_kernel(
    const float* __restrict__ x, const float* __restrict__ A,
    float* __restrict__ R, float* __restrict__ T)
{
  const int tid  = threadIdx.x;
  const int lane = tid & 31;
  const int wave = tid >> 5;
  const int rsel = lane & 15;     // matrix row (A-op) / col (B-op) select
  const int hi   = lane >> 4;     // half-wave: selects K sub-ranges
  const int rowBase = blockIdx.x * 128 + wave * 16;
  const int row = rowBase + rsel;

  const float* xr = x + (size_t)row  * Kdim;
  const float* ar = A + (size_t)rsel * Kdim;

  v8f c = {0.f,0.f,0.f,0.f,0.f,0.f,0.f,0.f};
  float rsum = 0.0f;

  for (int kk = 0; kk < Kdim; kk += 32) {
    // A-operand (x): lanes 0-15 hold K {0..7,16..23}, lanes 16-31 {8..15,24..31}
    const int k0 = kk + hi * 8;
    v4f xa = *(const v4f*)(xr + k0);
    v4f xb = *(const v4f*)(xr + k0 + 4);
    v4f xc = *(const v4f*)(xr + k0 + 16);
    v4f xd = *(const v4f*)(xr + k0 + 20);
    rsum += xa[0]+xa[1]+xa[2]+xa[3] + xb[0]+xb[1]+xb[2]+xb[3]
          + xc[0]+xc[1]+xc[2]+xc[3] + xd[0]+xd[1]+xd[2]+xd[3];
    V16H a; a.h[0] = cvt8(xa, xb); a.h[1] = cvt8(xc, xd);

    // B-operand (A^T): lane n<16 -> A[n][kk+{0..7,16..23}], lane>=16 -> A[n-16][kk+{8..15,24..31}]
    const float* ap = ar + kk + hi * 8;
    v4f b0 = *(const v4f*)(ap);
    v4f b1 = *(const v4f*)(ap + 4);
    v4f b2 = *(const v4f*)(ap + 16);
    v4f b3 = *(const v4f*)(ap + 20);
    V16H b; b.h[0] = cvt8(b0, b1); b.h[1] = cvt8(b2, b3);

    c = __builtin_amdgcn_wmma_f32_16x16x32_f16(
        false, a.v, false, b.v, (short)0, c, false, false);
  }

  // rowsum: lane l and lane l^16 hold complementary K subsets of the same row
  float other = __shfl_xor(rsum, 16, 32);
  if (hi == 0) R[row] = rsum + other;

  // D layout: M = v + 8*hi, N = rsel
  #pragma unroll
  for (int v = 0; v < 8; ++v)
    T[(size_t)(rowBase + hi * 8 + v) * RANK + rsel] = c[v];
}

// -------------------------------------------------------------------------
// Stage 2: P = x @ q^T in *full fp32* via V_WMMA_F32_16X16X4_F32.
//   x tile : async DMA global->LDS (double-buffered, ASYNCcnt)
//   q tile : int32 -> f32 convert once at stage time
// epilogue: out = scale[o]*P + min[o]*R[m] + bias[o] + 2*<T[m,:], B[o,:]>
// -------------------------------------------------------------------------
constexpr int BM = 128, BN = 128, BK = 32;
constexpr int FSTR = 36;   // floats per LDS row (32 + 4 pad, rows stay 16B aligned)

__global__ __launch_bounds__(256) void qlora_main_kernel(
    const float* __restrict__ x, const int* __restrict__ qw,
    const float* __restrict__ wmin, const float* __restrict__ wscale,
    const float* __restrict__ bias, const float* __restrict__ Bm,
    const float* __restrict__ R, const float* __restrict__ T,
    float* __restrict__ out)
{
  __shared__ float xs[2][BM * FSTR];   // double-buffered x tile (f32)
  __shared__ float qs[BN * FSTR];      // q tile (dequant-to-int value, f32)

  const int tid  = threadIdx.x;
  const int lane = tid & 31;
  const int wave = tid >> 5;
  const int rsel = lane & 15;
  const int hi   = lane >> 4;

  const int mBase = blockIdx.y * BM;
  const int nBase = blockIdx.x * BN;

  // 8 waves in a 2(M) x 4(N) grid; each wave owns 4x2 16x16 WMMA tiles
  const int waveM = (wave & 1) * 64;
  const int waveN = (wave >> 1) * 32;

  // cooperative tile load mapping: 2 threads per row, 16 f32 each
  const int lrow = tid >> 1;
  const int lcol = (tid & 1) * 16;

  v8f acc[4][2];
  #pragma unroll
  for (int i = 0; i < 4; ++i)
    #pragma unroll
    for (int j = 0; j < 2; ++j)
      acc[i][j] = (v8f){0.f,0.f,0.f,0.f,0.f,0.f,0.f,0.f};

  const float* xg = x  + (size_t)(mBase + lrow) * Kdim + lcol;
  const int*   qg = qw + (size_t)(nBase + lrow) * Kdim + lcol;
  const int lofs = lrow * FSTR + lcol;

#ifdef HAVE_ASYNC_LDS
  // prologue: kick off DMA of the first x tile into buffer 0
  #pragma unroll
  for (int j = 0; j < 4; ++j)
    ASYNC_CP_B128(xg + j * 4, &xs[0][lofs + j * 4]);
#endif

  for (int kk = 0; kk < Kdim; kk += BK) {
    const int buf = (kk / BK) & 1;

    // ---- q tile: global int32 -> VGPR -> f32 -> LDS -----------------
    v4i q0 = *(const v4i*)(qg + kk);
    v4i q1 = *(const v4i*)(qg + kk + 4);
    v4i q2 = *(const v4i*)(qg + kk + 8);
    v4i q3 = *(const v4i*)(qg + kk + 12);

    const bool more = (kk + BK) < Kdim;

#ifdef HAVE_ASYNC_LDS
    // issue next x tile's DMA into the other buffer before waiting
    if (more) {
      #pragma unroll
      for (int j = 0; j < 4; ++j)
        ASYNC_CP_B128(xg + kk + BK + j * 4, &xs[buf ^ 1][lofs + j * 4]);
    }
#else
    {
      v4f x0 = *(const v4f*)(xg + kk);
      v4f x1 = *(const v4f*)(xg + kk + 4);
      v4f x2 = *(const v4f*)(xg + kk + 8);
      v4f x3 = *(const v4f*)(xg + kk + 12);
      *(v4f*)(&xs[buf][lofs])      = x0;
      *(v4f*)(&xs[buf][lofs + 4])  = x1;
      *(v4f*)(&xs[buf][lofs + 8])  = x2;
      *(v4f*)(&xs[buf][lofs + 12]) = x3;
    }
#endif

    if (more) __builtin_prefetch(qg + kk + BK, 0, 1);   // global_prefetch_b8

    *(v4f*)(&qs[lofs])      = cvt4i(q0);
    *(v4f*)(&qs[lofs + 4])  = cvt4i(q1);
    *(v4f*)(&qs[lofs + 8])  = cvt4i(q2);
    *(v4f*)(&qs[lofs + 12]) = cvt4i(q3);

#ifdef HAVE_ASYNC_LDS
    // oldest 4 async copies (current buffer) must be complete; the 4 just
    // issued for the next buffer may remain in flight
    if (more) { WAIT_ASYNC(4); } else { WAIT_ASYNC(0); }
#endif
    __syncthreads();

    // ---- 8 K-steps of V_WMMA_F32_16X16X4_F32 ------------------------
    // f32 A-op 16x4: lanes 0-15 hold K{0,1}, lanes 16-31 K{2,3} (VGPR j = K 2*hi+j)
    // f32 B-op 4x16: same split, N = lane%16
    #pragma unroll
    for (int ks = 0; ks < 8; ++ks) {
      const int kb = ks * 4 + 2 * hi;
      v2f aop[4], bop[2];
      #pragma unroll
      for (int mi = 0; mi < 4; ++mi)
        aop[mi] = *(const v2f*)&xs[buf][(waveM + mi * 16 + rsel) * FSTR + kb];
      #pragma unroll
      for (int ni = 0; ni < 2; ++ni)
        bop[ni] = *(const v2f*)&qs[(waveN + ni * 16 + rsel) * FSTR + kb];
      #pragma unroll
      for (int mi = 0; mi < 4; ++mi)
        #pragma unroll
        for (int ni = 0; ni < 2; ++ni)
          acc[mi][ni] = __builtin_amdgcn_wmma_f32_16x16x4_f32(
              false, aop[mi], false, bop[ni], (short)0, acc[mi][ni],
              false, false);
    }
    __syncthreads();
  }

  // ---- epilogue: dequant affine + bias + rank-16 LoRA -------------------
  float sc[2], mn[2], bi[2];
  v4f bmv[2][4];
  #pragma unroll
  for (int ni = 0; ni < 2; ++ni) {
    const int col = nBase + waveN + ni * 16 + rsel;
    sc[ni] = wscale[col];
    mn[ni] = wmin[col];
    bi[ni] = bias[col];
    const v4f* bp = (const v4f*)(Bm + (size_t)col * RANK);
    bmv[ni][0] = bp[0]; bmv[ni][1] = bp[1]; bmv[ni][2] = bp[2]; bmv[ni][3] = bp[3];
  }

  #pragma unroll
  for (int mi = 0; mi < 4; ++mi) {
    const int row0 = mBase + waveM + mi * 16 + hi * 8;   // C/D layout: M = v + 8*hi
    #pragma unroll
    for (int v = 0; v < 8; ++v) {
      const int row = row0 + v;
      const float r = R[row];
      const v4f* tp = (const v4f*)(T + (size_t)row * RANK);
      v4f t0 = tp[0], t1 = tp[1], t2 = tp[2], t3 = tp[3];
      #pragma unroll
      for (int ni = 0; ni < 2; ++ni) {
        v4f p = t0 * bmv[ni][0] + t1 * bmv[ni][1] + t2 * bmv[ni][2] + t3 * bmv[ni][3];
        const float lora = p[0] + p[1] + p[2] + p[3];
        const int col = nBase + waveN + ni * 16 + rsel;
        out[(size_t)row * Ndim + col] =
            sc[ni] * acc[mi][ni][v] + mn[ni] * r + bi[ni] + 2.0f * lora;
      }
    }
  }
}

// -------------------------------------------------------------------------
extern "C" void kernel_launch(void* const* d_in, const int* in_sizes, int n_in,
                              void* d_out, int out_size, void* d_ws, size_t ws_size,
                              hipStream_t stream) {
  (void)in_sizes; (void)n_in; (void)out_size; (void)ws_size;
  const float* x      = (const float*)d_in[0];
  const int*   qw     = (const int*)  d_in[1];
  const float* wmin   = (const float*)d_in[2];
  const float* wscale = (const float*)d_in[3];
  const float* bias   = (const float*)d_in[4];
  const float* A      = (const float*)d_in[5];
  const float* Bm     = (const float*)d_in[6];
  float* out = (float*)d_out;

  float* R = (float*)d_ws;          // [Mdim]
  float* T = R + Mdim;              // [Mdim][RANK]

  qlora_stage1_kernel<<<dim3(Mdim / 128), 256, 0, stream>>>(x, A, R, T);
  qlora_main_kernel<<<dim3(Ndim / BN, Mdim / BM), 256, 0, stream>>>(
      x, qw, wmin, wscale, bias, Bm, R, T, out);
}
